// ImprovedGNNEncoder_32315333935105
// MI455X (gfx1250) — matrix operand order repaired and checked by reading, not compiled, not run on hardware
//
#include <hip/hip_runtime.h>
#include <hip/hip_bf16.h>

// ---------------------------------------------------------------------------
// GAT encoder for MI455X (gfx1250, wave32).
//  * Edge linear path algebraically folded: alpha_e = edge_attr @ q[l] + r[l]
//    (q = w_edge @ (w_edge_lin[l] . att_edge[l]))  -> kills the [E,128] GEMMs.
//  * Node GEMMs (x@w_node, h@w_lin[l]) run on v_wmma_f32_16x16x32_f16,
//    f16 operands / f32 accumulate, one 16x16 tile per wave, 8 waves/block.
//  * Gather/scatter of h_l (25.6 MB) stays resident in the 192 MB L2.
// ---------------------------------------------------------------------------

#define Nn   50000
#define Ee   500000
#define Gg   1000
#define NDd  64
#define EDd  16
#define Hh   128
#define HEADSh 4
#define CHc  32
#define EMBe 64

typedef __attribute__((ext_vector_type(16))) _Float16 v16h;
typedef __attribute__((ext_vector_type(8)))  float    v8f;

union ABu { v16h v; float4 f4[2]; };

// -------------------- WMMA GEMM:  C[M,Nc] = A[M,K](f16) * BT[Nc,K](f16) + bias
// M % 16 == 0, K % 32 == 0, Nc == 128.  grid.x = M/16, block = 256 (8 waves).
__global__ void k_wmma_gemm(const _Float16* __restrict__ A,
                            const _Float16* __restrict__ BT,
                            const float* __restrict__ bias,
                            float* __restrict__ C,
                            int K, int Nc) {
    const int lane = threadIdx.x & 31;
    const int wv   = threadIdx.x >> 5;
    const int row0 = blockIdx.x << 4;
    const int col0 = wv << 4;
    const int m    = lane & 15;
    const int ksA  = (lane >> 4) << 3;   // A: halves [ksA,ksA+8) and [ksA+16,ksA+24)
    const int ksB  = (lane >> 4) << 3;   // B: halves [ksB,ksB+8) and [ksB+16,ksB+24)

    const _Float16* arow = A  + (size_t)(row0 + m) * K + ksA;
    const _Float16* brow = BT + (size_t)(col0 + m) * K + ksB;

    v8f acc = {};
    for (int kk = 0; kk < K; kk += 32) {
        __builtin_prefetch(arow + kk + 64, 0, 0);   // global_prefetch_b8
        ABu a, b;
        a.f4[0] = *reinterpret_cast<const float4*>(arow + kk);
        a.f4[1] = *reinterpret_cast<const float4*>(arow + kk + 16);
        b.f4[0] = *reinterpret_cast<const float4*>(brow + kk);
        b.f4[1] = *reinterpret_cast<const float4*>(brow + kk + 16);
        acc = __builtin_amdgcn_wmma_f32_16x16x32_f16(
            /*neg_a=*/false, a.v, /*neg_b=*/false, b.v,
            /*c_mod=*/(short)0, acc, /*reuse_a=*/false, /*reuse_b=*/false);
    }
    const int rr = row0 + ((lane >> 4) << 3);
    const int cc = col0 + (lane & 15);
    const float bv = bias ? bias[cc] : 0.f;
#pragma unroll
    for (int r = 0; r < 8; ++r)
        C[(size_t)(rr + r) * Nc + cc] = acc[r] + bv;
}

// -------------------- small utility kernels --------------------------------
__global__ void k_zero(float* p, size_t n) {
    size_t i = (size_t)blockIdx.x * blockDim.x + threadIdx.x;
    if (i < n) p[i] = 0.f;
}
__global__ void k_convert_f16(const float* __restrict__ in, _Float16* __restrict__ out, size_t n) {
    size_t i = (size_t)blockIdx.x * blockDim.x + threadIdx.x;
    if (i < n) out[i] = (_Float16)in[i];
}
// in: [nmat][rows][cols] (f32) -> out: [nmat][cols][rows] (f16)
__global__ void k_transpose_f16(const float* __restrict__ in, _Float16* __restrict__ out,
                                int rows, int cols, int nmat) {
    size_t total = (size_t)nmat * rows * cols;
    size_t i = (size_t)blockIdx.x * blockDim.x + threadIdx.x;
    if (i >= total) return;
    int c = (int)(i % cols);
    size_t t = i / cols;
    int r = (int)(t % rows);
    int mm = (int)(t / rows);
    out[((size_t)mm * cols + c) * rows + r] = (_Float16)in[i];
}

// p[l][k][hd] = sum_c w[l][k][hd*32+c] * att[l][hd][c]     (w:[3][128][128])
__global__ void k_fold_att(const float* __restrict__ w, const float* __restrict__ att,
                           float* __restrict__ p) {
    int idx = blockIdx.x * blockDim.x + threadIdx.x;    // 3*128*4
    if (idx >= 3 * Hh * HEADSh) return;
    int hd = idx & 3, k = (idx >> 2) & 127, l = idx >> 9;
    const float* wr = w + ((size_t)l * Hh + k) * Hh + hd * CHc;
    const float* ar = att + ((size_t)l * HEADSh + hd) * CHc;
    float s = 0.f;
#pragma unroll
    for (int c = 0; c < CHc; ++c) s += wr[c] * ar[c];
    p[idx] = s;
}

// q[l][j][hd] = sum_k w_edge[j][k] * p_edge[l][k][hd];  r[l][hd] uses b_edge.
__global__ void k_fold_edge(const float* __restrict__ w_edge, const float* __restrict__ b_edge,
                            const float* __restrict__ p_edge,
                            float* __restrict__ q, float* __restrict__ r) {
    int idx = blockIdx.x * blockDim.x + threadIdx.x;    // 3*17*4
    if (idx >= 3 * 17 * 4) return;
    int hd = idx & 3, j = (idx >> 2) % 17, l = (idx >> 2) / 17;
    const float* p = p_edge + (size_t)l * Hh * 4;
    float s = 0.f;
    if (j < EDd) {
        const float* wr = w_edge + (size_t)j * Hh;
        for (int k = 0; k < Hh; ++k) s += wr[k] * p[k * 4 + hd];
        q[((size_t)l * EDd + j) * 4 + hd] = s;
    } else {
        for (int k = 0; k < Hh; ++k) s += b_edge[k] * p[k * 4 + hd];
        r[l * 4 + hd] = s;
    }
}

// alphaE[l][e][hd] = r[l][hd] + sum_j edge_attr[e][j] * q[l][j][hd]
__global__ void k_alpha_edge(const float* __restrict__ ea, const float* __restrict__ q,
                             const float* __restrict__ r, float* __restrict__ alphaE) {
    int e = blockIdx.x * blockDim.x + threadIdx.x;
    if (e >= Ee) return;
    float a[EDd];
    const float* row = ea + (size_t)e * EDd;
#pragma unroll
    for (int j = 0; j < EDd; ++j) a[j] = row[j];
#pragma unroll
    for (int l = 0; l < 3; ++l) {
        float s0 = r[l*4+0], s1 = r[l*4+1], s2 = r[l*4+2], s3 = r[l*4+3];
        const float* ql = q + (size_t)l * EDd * 4;
#pragma unroll
        for (int j = 0; j < EDd; ++j) {
            s0 += a[j] * ql[j*4+0]; s1 += a[j] * ql[j*4+1];
            s2 += a[j] * ql[j*4+2]; s3 += a[j] * ql[j*4+3];
        }
        float* o = alphaE + ((size_t)l * Ee + e) * 4;
        o[0] = s0; o[1] = s1; o[2] = s2; o[3] = s3;
    }
}

// per node: a_s / a_d dots against att vectors; init segment-softmax state
__global__ void k_node_att(const float* __restrict__ hl, const float* __restrict__ att_src,
                           const float* __restrict__ att_dst,
                           float* __restrict__ a_s, float* __restrict__ a_d,
                           float* __restrict__ mx, float* __restrict__ den) {
    int n = blockIdx.x * blockDim.x + threadIdx.x;
    if (n >= Nn) return;
    const float* row = hl + (size_t)n * Hh;
#pragma unroll
    for (int hd = 0; hd < HEADSh; ++hd) {
        const float* as = att_src + hd * CHc;
        const float* ad = att_dst + hd * CHc;
        float ss = 0.f, sd = 0.f;
#pragma unroll
        for (int c = 0; c < CHc; ++c) {
            float v = row[hd * CHc + c];
            ss += v * as[c]; sd += v * ad[c];
        }
        a_s[n*4+hd] = ss; a_d[n*4+hd] = sd;
        mx[n*4+hd] = -3.0e38f; den[n*4+hd] = 0.f;
    }
}

// loop-free monotone float atomic max (sign-split encoding)
__device__ inline void atomicMaxFloat(float* addr, float val) {
    if (val >= 0.f) atomicMax((int*)addr, __float_as_int(val));
    else            atomicMin((unsigned int*)addr, (unsigned int)__float_as_int(val));
}

__global__ void k_edge_alpha(const long long* __restrict__ src, const long long* __restrict__ dst,
                             const float* __restrict__ a_s, const float* __restrict__ a_d,
                             const float* __restrict__ aE, float* __restrict__ alpha,
                             float* __restrict__ mx) {
    int e = blockIdx.x * blockDim.x + threadIdx.x;
    if (e >= Ee) return;
    long long s = src[e], d = dst[e];
#pragma unroll
    for (int hd = 0; hd < 4; ++hd) {
        float v = a_s[s*4+hd] + a_d[d*4+hd] + aE[(size_t)e*4+hd];
        v = (v > 0.f) ? v : 0.2f * v;                 // leaky_relu(0.2)
        alpha[(size_t)e*4+hd] = v;
        atomicMaxFloat(&mx[d*4+hd], v);
    }
}

__global__ void k_edge_exp(const long long* __restrict__ dst, float* __restrict__ alpha,
                           const float* __restrict__ mx, float* __restrict__ den) {
    int e = blockIdx.x * blockDim.x + threadIdx.x;
    if (e >= Ee) return;
    long long d = dst[e];
#pragma unroll
    for (int hd = 0; hd < 4; ++hd) {
        float ex = expf(alpha[(size_t)e*4+hd] - mx[d*4+hd]);
        alpha[(size_t)e*4+hd] = ex;
        atomicAdd(&den[d*4+hd], ex);
    }
}

// one wave per edge: lane handles 4 consecutive channels (head = lane>>3)
__global__ void k_edge_scatter(const long long* __restrict__ src, const long long* __restrict__ dst,
                               const float* __restrict__ ex, const float* __restrict__ den,
                               const float* __restrict__ hl, float* __restrict__ hnew) {
    int t = blockIdx.x * blockDim.x + threadIdx.x;
    int e = t >> 5, lane = t & 31;
    if (e >= Ee) return;
    long long s = src[e], d = dst[e];
    int hd = lane >> 3;
    float w = ex[(size_t)e*4+hd] / (den[d*4+hd] + 1e-16f);
    const float4 v = *reinterpret_cast<const float4*>(hl + (size_t)s * Hh + lane * 4);
    float* o = hnew + (size_t)d * Hh + lane * 4;
    atomicAdd(o+0, v.x * w); atomicAdd(o+1, v.y * w);
    atomicAdd(o+2, v.z * w); atomicAdd(o+3, v.w * w);
}

// bias + relu + eval-mode BN, write f32 h and f16 copy for next GEMM
__global__ void k_post(const float* __restrict__ hnew, const float* __restrict__ cb,
                       const float* __restrict__ g, const float* __restrict__ b,
                       const float* __restrict__ mu, const float* __restrict__ var,
                       float* __restrict__ h, _Float16* __restrict__ h16) {
    size_t i = (size_t)blockIdx.x * blockDim.x + threadIdx.x;
    if (i >= (size_t)Nn * Hh) return;
    int k = (int)(i & (Hh - 1));
    float v = hnew[i] + cb[k];
    v = fmaxf(v, 0.f);
    v = (v - mu[k]) * (g[k] * rsqrtf(var[k] + 1e-5f)) + b[k];
    h[i] = v; h16[i] = (_Float16)v;
}

// one wave per node: segment-sum into pooled[batch], counts
__global__ void k_pool(const float* __restrict__ h, const long long* __restrict__ batch,
                       float* __restrict__ pooled, float* __restrict__ cnt) {
    int t = blockIdx.x * blockDim.x + threadIdx.x;
    int n = t >> 5, lane = t & 31;
    if (n >= Nn) return;
    long long bi = batch[n];
    const float4 v = *reinterpret_cast<const float4*>(h + (size_t)n * Hh + lane * 4);
    float* o = pooled + (size_t)bi * Hh + lane * 4;
    atomicAdd(o+0, v.x); atomicAdd(o+1, v.y);
    atomicAdd(o+2, v.z); atomicAdd(o+3, v.w);
    if (lane == 0) atomicAdd(&cnt[bi], 1.f);
}

__global__ void k_head(const float* __restrict__ pooled, const float* __restrict__ cnt,
                       const float* __restrict__ w_out, const float* __restrict__ b_out,
                       float* __restrict__ out) {
    int t = blockIdx.x * blockDim.x + threadIdx.x;
    if (t >= Gg * EMBe) return;
    int g = t >> 6, j = t & 63;
    float inv = 1.f / fmaxf(cnt[g], 1.f);
    const float* pr = pooled + (size_t)g * Hh;
    float s = 0.f;
#pragma unroll 4
    for (int k = 0; k < Hh; ++k) s += pr[k] * w_out[k * EMBe + j];
    out[t] = s * inv + b_out[j];
}

// ---------------------------------------------------------------------------
static inline void* bump(char*& p, size_t bytes) {
    void* r = p;
    p += (bytes + 255) & ~(size_t)255;
    return r;
}
static inline dim3 gr(size_t n) { return dim3((unsigned)((n + 255) / 256)); }

extern "C" void kernel_launch(void* const* d_in, const int* in_sizes, int n_in,
                              void* d_out, int out_size, void* d_ws, size_t ws_size,
                              hipStream_t stream) {
    const float*     x         = (const float*)d_in[0];
    const float*     edge_attr = (const float*)d_in[1];
    const long long* ei        = (const long long*)d_in[2];
    const long long* batch     = (const long long*)d_in[3];
    const float*     w_node    = (const float*)d_in[4];
    const float*     b_node    = (const float*)d_in[5];
    const float*     w_edge    = (const float*)d_in[6];
    const float*     b_edge    = (const float*)d_in[7];
    const float*     w_lin     = (const float*)d_in[8];
    const float*     w_edge_lin= (const float*)d_in[9];
    const float*     att_src   = (const float*)d_in[10];
    const float*     att_dst   = (const float*)d_in[11];
    const float*     att_edge  = (const float*)d_in[12];
    const float*     conv_bias = (const float*)d_in[13];
    const float*     bn_gamma  = (const float*)d_in[14];
    const float*     bn_beta   = (const float*)d_in[15];
    const float*     bn_mean   = (const float*)d_in[16];
    const float*     bn_var    = (const float*)d_in[17];
    const float*     w_out     = (const float*)d_in[18];
    const float*     b_out     = (const float*)d_in[19];
    float*           out       = (float*)d_out;

    const long long* src = ei;
    const long long* dst = ei + Ee;

    char* ws = (char*)d_ws;
    _Float16* x16    = (_Float16*)bump(ws, (size_t)Nn * NDd * 2);
    _Float16* wnodeT = (_Float16*)bump(ws, (size_t)Hh * NDd * 2);
    _Float16* wlinT  = (_Float16*)bump(ws, (size_t)3 * Hh * Hh * 2);
    float* p_edge = (float*)bump(ws, (size_t)3 * Hh * 4 * 4);
    float* q_edge = (float*)bump(ws, (size_t)3 * EDd * 4 * 4);
    float* r_edge = (float*)bump(ws, (size_t)3 * 4 * 4);
    float* alphaE = (float*)bump(ws, (size_t)3 * Ee * 4 * 4);
    float* h      = (float*)bump(ws, (size_t)Nn * Hh * 4);
    _Float16* h16 = (_Float16*)bump(ws, (size_t)Nn * Hh * 2);
    float* hl     = (float*)bump(ws, (size_t)Nn * Hh * 4);
    float* hnew   = (float*)bump(ws, (size_t)Nn * Hh * 4);
    float* a_s    = (float*)bump(ws, (size_t)Nn * 4 * 4);
    float* a_d    = (float*)bump(ws, (size_t)Nn * 4 * 4);
    float* mx     = (float*)bump(ws, (size_t)Nn * 4 * 4);
    float* den    = (float*)bump(ws, (size_t)Nn * 4 * 4);
    float* alpha  = (float*)bump(ws, (size_t)Ee * 4 * 4);
    float* pooled = (float*)bump(ws, (size_t)Gg * Hh * 4);
    float* cnt    = (float*)bump(ws, (size_t)Gg * 4);
    // total ~132 MB of d_ws

    // ---- one-time transforms -------------------------------------------------
    k_transpose_f16<<<gr((size_t)NDd * Hh), 256, 0, stream>>>(w_node, wnodeT, NDd, Hh, 1);
    k_transpose_f16<<<gr((size_t)3 * Hh * Hh), 256, 0, stream>>>(w_lin, wlinT, Hh, Hh, 3);
    k_convert_f16<<<gr((size_t)Nn * NDd), 256, 0, stream>>>(x, x16, (size_t)Nn * NDd);
    k_fold_att<<<gr(3 * Hh * 4), 256, 0, stream>>>(w_edge_lin, att_edge, p_edge);
    k_fold_edge<<<1, 256, 0, stream>>>(w_edge, b_edge, p_edge, q_edge, r_edge);
    k_alpha_edge<<<gr(Ee), 256, 0, stream>>>(edge_attr, q_edge, r_edge, alphaE);

    // ---- node embedding: h = x @ w_node + b_node (WMMA, K=64) ---------------
    k_wmma_gemm<<<Nn / 16, 256, 0, stream>>>(x16, wnodeT, b_node, h, NDd, Hh);
    k_convert_f16<<<gr((size_t)Nn * Hh), 256, 0, stream>>>(h, h16, (size_t)Nn * Hh);

    // ---- 3 GAT layers --------------------------------------------------------
    for (int l = 0; l < 3; ++l) {
        k_wmma_gemm<<<Nn / 16, 256, 0, stream>>>(h16, wlinT + (size_t)l * Hh * Hh,
                                                 nullptr, hl, Hh, Hh);
        k_node_att<<<gr(Nn), 256, 0, stream>>>(hl, att_src + l * 128, att_dst + l * 128,
                                               a_s, a_d, mx, den);
        k_zero<<<gr((size_t)Nn * Hh), 256, 0, stream>>>(hnew, (size_t)Nn * Hh);
        k_edge_alpha<<<gr(Ee), 256, 0, stream>>>(src, dst, a_s, a_d,
                                                 alphaE + (size_t)l * Ee * 4, alpha, mx);
        k_edge_exp<<<gr(Ee), 256, 0, stream>>>(dst, alpha, mx, den);
        k_edge_scatter<<<(Ee * 32) / 256, 256, 0, stream>>>(src, dst, alpha, den, hl, hnew);
        k_post<<<gr((size_t)Nn * Hh), 256, 0, stream>>>(hnew, conv_bias + l * 128,
                                                        bn_gamma + l * 128, bn_beta + l * 128,
                                                        bn_mean + l * 128, bn_var + l * 128,
                                                        h, h16);
    }

    // ---- global mean pool + head --------------------------------------------
    k_zero<<<gr((size_t)Gg * Hh), 256, 0, stream>>>(pooled, (size_t)Gg * Hh);
    k_zero<<<gr(Gg), 256, 0, stream>>>(cnt, Gg);
    k_pool<<<(Nn * 32) / 256, 256, 0, stream>>>(h, batch, pooled, cnt);
    k_head<<<gr((size_t)Gg * EMBe), 256, 0, stream>>>(pooled, cnt, w_out, b_out, out);
}
